// RealImagTernaryLinear_73555609911439
// MI455X (gfx1250) — compile-verified, matrix operand order checked by reading, not run on hardware
//
#include <hip/hip_runtime.h>

#define IN_F   4096
#define OUT_F  4096
#define TOKENS 8192
#define KSTEP  32
#define NKT    (IN_F / KSTEP)

typedef __attribute__((ext_vector_type(2)))  __bf16 v2bf;
typedef __attribute__((ext_vector_type(8)))  __bf16 v8bf;
typedef __attribute__((ext_vector_type(16))) __bf16 v16bf;
typedef __attribute__((ext_vector_type(8)))  float  v8f;

// Pack two f32 -> two bf16 (RNE) in one 32-bit word. Exact for {-1,0,1}.
__device__ __forceinline__ unsigned pack2bf(float f0, float f1) {
#if __has_builtin(__builtin_amdgcn_cvt_pk_bf16_f32)
    v2bf v = __builtin_amdgcn_cvt_pk_bf16_f32(f0, f1);
    return __builtin_bit_cast(unsigned, v);
#else
    unsigned u0 = __builtin_bit_cast(unsigned, f0);
    unsigned u1 = __builtin_bit_cast(unsigned, f1);
    u0 += 0x7FFFu + ((u0 >> 16) & 1u);
    u1 += 0x7FFFu + ((u1 >> 16) & 1u);
    // result = {u1.hi16, u0.hi16}: byte0<-u0.b2, byte1<-u0.b3, byte2<-u1.b2, byte3<-u1.b3
    return __builtin_amdgcn_perm(u1, u0, 0x07060302u);
#endif
}

__device__ __forceinline__ float tern(float v, float sc) {
    return fminf(1.f, fmaxf(-1.f, rintf(v * sc)));
}

// Load one wave's 16x32 bf16 fragment from a row-major [row][32] LDS tile.
// ISA 16-bit A layout: lanes 0-15 hold K=0..7 / K=16..23; lanes 16-31 hold
// K=8..15 / K=24..31 of the same 16 rows. B ([N][K] row-major) is identical.
__device__ __forceinline__ v16bf frag(const __bf16* s, int row, int lane) {
    const int koff = (lane >> 4) * 8;
    v8bf lo = *(const v8bf*)(s + row * 32 + koff);
    v8bf hi = *(const v8bf*)(s + row * 32 + koff + 16);
    return __builtin_shufflevector(lo, hi, 0,1,2,3,4,5,6,7,8,9,10,11,12,13,14,15);
}

__global__ void zero2_kernel(float* ws) {
    if (threadIdx.x < 2) ws[threadIdx.x] = 0.0f;
}

__global__ __launch_bounds__(256)
void abssum_kernel(const float* __restrict__ w0, const float* __restrict__ w1,
                   float* __restrict__ ws) {
    const float* w = blockIdx.y ? w1 : w0;
    const int n = IN_F * OUT_F;
    float s = 0.0f;
    for (int i = blockIdx.x * 256 + threadIdx.x; i < n; i += gridDim.x * 256)
        s += fabsf(w[i]);
    __shared__ float red[256];
    red[threadIdx.x] = s;
    __syncthreads();
    for (int off = 128; off > 0; off >>= 1) {
        if (threadIdx.x < off) red[threadIdx.x] += red[threadIdx.x + off];
        __syncthreads();
    }
    if (threadIdx.x == 0) atomicAdd(&ws[blockIdx.y], red[0]);
}

// Complex ternary GEMM: block = 64(M) x 64(N), 8 waves, wave tile = 16(M) x 32(N).
// Double-buffered LDS stage (32 KB), one barrier per K-step, global prefetch of
// the next tile overlapped with the current tile's WMMAs.
__global__ __launch_bounds__(256)
void cgemm_tern_kernel(const float* __restrict__ xre, const float* __restrict__ xim,
                       const float* __restrict__ wre, const float* __restrict__ wim,
                       const float* __restrict__ sums, float* __restrict__ out) {
    __shared__ __bf16 sbuf[2][4][64 * 32];   // [buffer][Are,Aim,Bre,Bim]

    const float inv_n = 1.0f / ((float)IN_F * (float)OUT_F);
    const float dR = fmaxf(sums[0] * inv_n, 1e-5f);   // Delta = max(mean|w|, 1e-5)
    const float dI = fmaxf(sums[1] * inv_n, 1e-5f);
    const float sR = 1.0f / dR;                        // quantization scale
    const float sI = 1.0f / dI;

    const int t    = threadIdx.x;
    const int lane = t & 31;
    const int wv   = t >> 5;          // 0..7
    const int m0   = (wv & 3) * 16;   // wave M sub-tile in block
    const int n0   = (wv >> 2) * 32;  // wave N base (2 sub-tiles of 16)

    const int mblk = blockIdx.x * 64;
    const int nblk = blockIdx.y * 64;

    const int lr = t >> 2;            // loader row 0..63
    const int lk = (t & 3) * 8;       // loader K offset 0/8/16/24

    const float* src0 = xre + (size_t)(mblk + lr) * IN_F + lk;
    const float* src1 = xim + (size_t)(mblk + lr) * IN_F + lk;
    const float* src2 = wre + (size_t)(nblk + lr) * IN_F + lk;
    const float* src3 = wim + (size_t)(nblk + lr) * IN_F + lk;
    const int lds_off = lr * 32 + lk;

    float4 r[8];  // staged registers: 2 float4 per matrix

    auto gload = [&](int k0) {
        r[0] = ((const float4*)(src0 + k0))[0];  r[1] = ((const float4*)(src0 + k0))[1];
        r[2] = ((const float4*)(src1 + k0))[0];  r[3] = ((const float4*)(src1 + k0))[1];
        r[4] = ((const float4*)(src2 + k0))[0];  r[5] = ((const float4*)(src2 + k0))[1];
        r[6] = ((const float4*)(src3 + k0))[0];  r[7] = ((const float4*)(src3 + k0))[1];
    };

    auto stage = [&](int buf) {
        uint4 p;
        // x_re
        p.x = pack2bf(r[0].x, r[0].y);  p.y = pack2bf(r[0].z, r[0].w);
        p.z = pack2bf(r[1].x, r[1].y);  p.w = pack2bf(r[1].z, r[1].w);
        *(uint4*)&sbuf[buf][0][lds_off] = p;
        // x_im
        p.x = pack2bf(r[2].x, r[2].y);  p.y = pack2bf(r[2].z, r[2].w);
        p.z = pack2bf(r[3].x, r[3].y);  p.w = pack2bf(r[3].z, r[3].w);
        *(uint4*)&sbuf[buf][1][lds_off] = p;
        // w_re -> ternary
        p.x = pack2bf(tern(r[4].x, sR), tern(r[4].y, sR));
        p.y = pack2bf(tern(r[4].z, sR), tern(r[4].w, sR));
        p.z = pack2bf(tern(r[5].x, sR), tern(r[5].y, sR));
        p.w = pack2bf(tern(r[5].z, sR), tern(r[5].w, sR));
        *(uint4*)&sbuf[buf][2][lds_off] = p;
        // w_im -> ternary
        p.x = pack2bf(tern(r[6].x, sI), tern(r[6].y, sI));
        p.y = pack2bf(tern(r[6].z, sI), tern(r[6].w, sI));
        p.z = pack2bf(tern(r[7].x, sI), tern(r[7].y, sI));
        p.w = pack2bf(tern(r[7].z, sI), tern(r[7].w, sI));
        *(uint4*)&sbuf[buf][3][lds_off] = p;
    };

    v8f acc_rr[2] = {}; // x_re @ q_re^T
    v8f acc_ii[2] = {}; // x_im @ q_im^T
    v8f acc_ri[2] = {}; // x_re @ q_im^T
    v8f acc_ir[2] = {}; // x_im @ q_re^T

    // prologue: fill buffer 0
    gload(0);
    stage(0);
    __syncthreads();

    const int ar = m0 + (lane & 15);
    for (int kt = 0; kt < NKT; ++kt) {
        const int cur = kt & 1;

        // issue next tile's global loads early (latency hidden under WMMAs)
        if (kt + 1 < NKT) gload((kt + 1) * KSTEP);

        // compute from current buffer
        v16bf a_re = frag(sbuf[cur][0], ar, lane);
        v16bf a_im = frag(sbuf[cur][1], ar, lane);
#pragma unroll
        for (int j = 0; j < 2; ++j) {
            const int br = n0 + j * 16 + (lane & 15);
            v16bf b_re = frag(sbuf[cur][2], br, lane);
            v16bf b_im = frag(sbuf[cur][3], br, lane);
            acc_rr[j] = __builtin_amdgcn_wmma_f32_16x16x32_bf16(
                false, a_re, false, b_re, (short)0, acc_rr[j], false, false);
            acc_ii[j] = __builtin_amdgcn_wmma_f32_16x16x32_bf16(
                false, a_im, false, b_im, (short)0, acc_ii[j], false, false);
            acc_ri[j] = __builtin_amdgcn_wmma_f32_16x16x32_bf16(
                false, a_re, false, b_im, (short)0, acc_ri[j], false, false);
            acc_ir[j] = __builtin_amdgcn_wmma_f32_16x16x32_bf16(
                false, a_im, false, b_re, (short)0, acc_ir[j], false, false);
        }

        // convert + store next tile into the other buffer
        if (kt + 1 < NKT) stage(cur ^ 1);
        __syncthreads();
    }

    // ---- epilogue: apply dequant scales, write y_re / y_im ----
    float* out_re = out;
    float* out_im = out + (size_t)TOKENS * OUT_F;
    const int mrow = mblk + m0 + (lane >> 4) * 8;  // C layout: VGPR g -> M=g (+8 for lanes 16-31)
#pragma unroll
    for (int j = 0; j < 2; ++j) {
        const int ncol = nblk + n0 + j * 16 + (lane & 15);
#pragma unroll
        for (int g = 0; g < 8; ++g) {
            const size_t idx = (size_t)(mrow + g) * OUT_F + ncol;
            out_re[idx] = dR * acc_rr[j][g] - dI * acc_ii[j][g];
            out_im[idx] = dI * acc_ri[j][g] + dR * acc_ir[j][g];
        }
    }
}

extern "C" void kernel_launch(void* const* d_in, const int* in_sizes, int n_in,
                              void* d_out, int out_size, void* d_ws, size_t ws_size,
                              hipStream_t stream) {
    const float* xre = (const float*)d_in[0];
    const float* xim = (const float*)d_in[1];
    const float* wre = (const float*)d_in[2];
    const float* wim = (const float*)d_in[3];
    float* ws  = (float*)d_ws;
    float* out = (float*)d_out;

    zero2_kernel<<<1, 32, 0, stream>>>(ws);
    abssum_kernel<<<dim3(512, 2), 256, 0, stream>>>(wre, wim, ws);
    cgemm_tern_kernel<<<dim3(TOKENS / 64, OUT_F / 64), 256, 0, stream>>>(
        xre, xim, wre, wim, ws, out);
}
